// TetraPermuter_41936060678663
// MI455X (gfx1250) — compile-verified, matrix operand order checked by reading, not run on hardware
//
#include <hip/hip_runtime.h>
#include <math.h>

typedef __attribute__((ext_vector_type(16))) __bf16 v16bf;
typedef __attribute__((ext_vector_type(8)))  float  v8f;

struct Q2 { uint4 a, b; };

static __device__ __forceinline__ v16bf make_frag(uint4 lo, uint4 hi) {
    Q2 q; q.a = lo; q.b = hi;
    return __builtin_bit_cast(v16bf, q);
}

// round-to-nearest-even f32 -> bf16 (as raw u16)
static __device__ __forceinline__ unsigned short f2bf(float f) {
    unsigned u = __builtin_bit_cast(unsigned, f);
    unsigned r = u + 0x7fffu + ((u >> 16) & 1u);
    return (unsigned short)(r >> 16);
}

// pack 4 f32 -> 4 bf16 (vectorized weight pre-conversion into workspace)
__global__ void cvt_f32_bf16_pk4(const float4* __restrict__ src,
                                 uint2* __restrict__ dst, int n4) {
    int i = blockIdx.x * blockDim.x + threadIdx.x;
    if (i >= n4) return;
    float4 v = src[i];
    uint2 o;
    o.x = (unsigned)f2bf(v.x) | ((unsigned)f2bf(v.y) << 16);
    o.y = (unsigned)f2bf(v.z) | ((unsigned)f2bf(v.w) << 16);
    dst[i] = o;
}

#define HH 256

// Fully fused: per block = 16 rows of N. 16 waves, each owns a 16-wide column slice.
__global__ __launch_bounds__(512) void tetra_fused(
    const float* __restrict__ x,       // (N,4,256)
    const float* __restrict__ bb,      // (4,256)
    const float* __restrict__ b1v,     // (256)
    const float* __restrict__ gammav,  // (256)
    const float* __restrict__ betav,   // (256)
    const float* __restrict__ b2v,     // (256)
    const int*   __restrict__ perms,   // (12,4)
    const unsigned short* __restrict__ WbB,  // bf16 (4,256,256)
    const unsigned short* __restrict__ W1B,  // bf16 (256,256)
    const unsigned short* __restrict__ W2B,  // bf16 (256,256)
    float* __restrict__ out)           // (N,256)
{
    __shared__ __align__(16) unsigned short lds_x[4 * 16 * HH]; // [(m*4+c)*256 + h]
    __shared__ __align__(16) unsigned short lds_act[16 * HH];   // [m*256 + h]

    const int tid  = threadIdx.x;
    const int wave = tid >> 5;
    const int lane = tid & 31;
    const int n0   = blockIdx.x << 4;

    // ---- cooperative load: x tile (16 rows x 4 corners x 256) fp32 -> bf16 LDS ----
    {
        const float4* xg = (const float4*)(x + (size_t)n0 * 4 * HH);
        uint2* lx = (uint2*)lds_x;
        #pragma unroll
        for (int j = 0; j < 8; ++j) {
            int i = tid + j * 512;       // 4096 float4 total
            float4 v = xg[i];
            uint2 o;
            o.x = (unsigned)f2bf(v.x) | ((unsigned)f2bf(v.y) << 16);
            o.y = (unsigned)f2bf(v.z) | ((unsigned)f2bf(v.w) << 16);
            lx[i] = o;
        }
    }

    // permutation table (uniform, unrolled -> scalar loads)
    int pm[12][4];
    #pragma unroll
    for (int p = 0; p < 12; ++p)
        #pragma unroll
        for (int k = 0; k < 4; ++k)
            pm[p][k] = perms[p * 4 + k];

    __syncthreads();

    const int g0    = wave << 4;
    const int col   = lane & 15;
    const int hsel  = (lane >> 4) << 3;  // K sub-base: lanes 16-31 carry K base+8
    const int mA    = lane & 15;         // A-fragment row for this lane
    const int g     = g0 + col;          // this lane's output column
    const int rbase = (lane >> 4) << 3;  // C/D row offset: lanes 16-31 -> rows 8..15

    // ---- stage 1: Y[k][c] = x_c @ Wb_k^T  (16x16 tiles, K=256 in steps of 32) ----
    v8f acc[4][4];
    #pragma unroll
    for (int k = 0; k < 4; ++k)
        #pragma unroll
        for (int c = 0; c < 4; ++c)
            acc[k][c] = v8f{0.f, 0.f, 0.f, 0.f, 0.f, 0.f, 0.f, 0.f};

    #pragma unroll
    for (int h0 = 0; h0 < HH; h0 += 32) {
        v16bf afr[4], bfr[4];
        #pragma unroll
        for (int c = 0; c < 4; ++c) {
            const unsigned short* ap = &lds_x[(mA * 4 + c) * HH + h0 + hsel];
            afr[c] = make_frag(*(const uint4*)ap, *(const uint4*)(ap + 16));
        }
        #pragma unroll
        for (int k = 0; k < 4; ++k) {
            const unsigned short* wp = WbB + ((size_t)k << 16) + (size_t)g * HH + h0 + hsel;
            bfr[k] = make_frag(*(const uint4*)wp, *(const uint4*)(wp + 16));
        }
        #pragma unroll
        for (int k = 0; k < 4; ++k)
            #pragma unroll
            for (int c = 0; c < 4; ++c)
                acc[k][c] = __builtin_amdgcn_wmma_f32_16x16x32_bf16(
                    false, afr[c], false, bfr[k], (short)0, acc[k][c], false, false);
    }

    // bias + tanh (elementwise on the 16 accumulators)
    float bbk[4];
    #pragma unroll
    for (int k = 0; k < 4; ++k) bbk[k] = bb[k * HH + g];
    #pragma unroll
    for (int k = 0; k < 4; ++k)
        #pragma unroll
        for (int c = 0; c < 4; ++c)
            #pragma unroll
            for (int r = 0; r < 8; ++r)
                acc[k][c][r] = tanhf(acc[k][c][r] + bbk[k]);

    // perm combine: nei = sum_p relu(sum_k T[k][perm[p][k]]);  fold in /3
    float nei[8];
    #pragma unroll
    for (int r = 0; r < 8; ++r) {
        float s = 0.f;
        #pragma unroll
        for (int p = 0; p < 12; ++p) {
            float t = acc[0][pm[p][0]][r] + acc[1][pm[p][1]][r]
                    + acc[2][pm[p][2]][r] + acc[3][pm[p][3]][r];
            s += fmaxf(t, 0.f);
        }
        nei[r] = s * (1.f / 3.f);
    }

    #pragma unroll
    for (int r = 0; r < 8; ++r)
        lds_act[(r + rbase) * HH + g] = f2bf(nei[r]);

    __syncthreads();

    // ---- stage 2: z = nei @ W1^T ; BN-ish scale + ReLU ----
    v8f zacc = v8f{0.f, 0.f, 0.f, 0.f, 0.f, 0.f, 0.f, 0.f};
    #pragma unroll
    for (int h0 = 0; h0 < HH; h0 += 32) {
        const unsigned short* ap = &lds_act[mA * HH + h0 + hsel];
        v16bf a = make_frag(*(const uint4*)ap, *(const uint4*)(ap + 16));
        const unsigned short* wp = W1B + (size_t)g * HH + h0 + hsel;
        v16bf b = make_frag(*(const uint4*)wp, *(const uint4*)(wp + 16));
        zacc = __builtin_amdgcn_wmma_f32_16x16x32_bf16(
            false, a, false, b, (short)0, zacc, false, false);
    }

    const float inv = 0.9999950000374997f;  // 1/sqrt(1 + 1e-5)
    const float ga = gammav[g] * inv, be = betav[g], bo = b1v[g];
    float hv[8];
    #pragma unroll
    for (int r = 0; r < 8; ++r) {
        float z = zacc[r] + bo;
        hv[r] = fmaxf(z * ga + be, 0.f);
    }

    __syncthreads();   // all reads of nei done before overwriting lds_act
    #pragma unroll
    for (int r = 0; r < 8; ++r)
        lds_act[(r + rbase) * HH + g] = f2bf(hv[r]);
    __syncthreads();

    // ---- stage 3: out = h @ W2^T + b2 ----
    v8f oacc = v8f{0.f, 0.f, 0.f, 0.f, 0.f, 0.f, 0.f, 0.f};
    #pragma unroll
    for (int h0 = 0; h0 < HH; h0 += 32) {
        const unsigned short* ap = &lds_act[mA * HH + h0 + hsel];
        v16bf a = make_frag(*(const uint4*)ap, *(const uint4*)(ap + 16));
        const unsigned short* wp = W2B + (size_t)g * HH + h0 + hsel;
        v16bf b = make_frag(*(const uint4*)wp, *(const uint4*)(wp + 16));
        oacc = __builtin_amdgcn_wmma_f32_16x16x32_bf16(
            false, a, false, b, (short)0, oacc, false, false);
    }

    const float b2g = b2v[g];
    #pragma unroll
    for (int r = 0; r < 8; ++r)
        out[(size_t)(n0 + r + rbase) * HH + g] = oacc[r] + b2g;
}

extern "C" void kernel_launch(void* const* d_in, const int* in_sizes, int n_in,
                              void* d_out, int out_size, void* d_ws, size_t ws_size,
                              hipStream_t stream) {
    (void)in_sizes; (void)n_in; (void)out_size; (void)ws_size;

    const float* x     = (const float*)d_in[0];
    const float* Wb    = (const float*)d_in[1];
    const float* bb    = (const float*)d_in[2];
    const float* W1    = (const float*)d_in[3];
    const float* b1    = (const float*)d_in[4];
    const float* gamma = (const float*)d_in[5];
    const float* beta  = (const float*)d_in[6];
    const float* W2    = (const float*)d_in[7];
    const float* b2    = (const float*)d_in[8];
    const int*   perms = (const int*)d_in[9];
    float* out = (float*)d_out;

    // workspace: bf16 weights [Wb: 4*65536][W1: 65536][W2: 65536]
    unsigned short* WbB = (unsigned short*)d_ws;
    unsigned short* W1B = WbB + 4 * 65536;
    unsigned short* W2B = W1B + 65536;

    const int nWb4 = (4 * 65536) / 4, nW4 = 65536 / 4;
    cvt_f32_bf16_pk4<<<(nWb4 + 255) / 256, 256, 0, stream>>>(
        (const float4*)Wb, (uint2*)WbB, nWb4);
    cvt_f32_bf16_pk4<<<(nW4 + 255) / 256, 256, 0, stream>>>(
        (const float4*)W1, (uint2*)W1B, nW4);
    cvt_f32_bf16_pk4<<<(nW4 + 255) / 256, 256, 0, stream>>>(
        (const float4*)W2, (uint2*)W2B, nW4);

    tetra_fused<<<65536 / 16, 512, 0, stream>>>(
        x, bb, b1, gamma, beta, b2, perms, WbB, W1B, W2B, out);
}
